// NMS_Export_73804718014593
// MI455X (gfx1250) — matrix operand order repaired and checked by reading, not compiled, not run on hardware
//
#include <hip/hip_runtime.h>
#include <stdint.h>

#define NB 16
#define NP 25200
#define NCOL 57
#define PAD 32768
#define TOPK 1024
#define MAXDET 300
#define CONF_T 0.001f
#define IOU_T 0.45f

typedef unsigned int v4u __attribute__((ext_vector_type(4)));
typedef int          v8i __attribute__((ext_vector_type(8)));
typedef int          v4i __attribute__((ext_vector_type(4)));
typedef __attribute__((ext_vector_type(16))) _Float16 v16h;
typedef __attribute__((ext_vector_type(8)))  float    v8f;

__global__ void zero_out(float* __restrict__ out, int n) {
    int i = blockIdx.x * blockDim.x + threadIdx.x;
    if (i < n) out[i] = 0.0f;
}

// Phase 1: streaming score computation. NC==1 -> conf = obj * cls0, class j == 0.
__global__ void score_kernel(const float* __restrict__ pred,
                             float* __restrict__ scores, int* __restrict__ sidx) {
    int g = blockIdx.x * blockDim.x + threadIdx.x;
    if (g >= NB * PAD) return;
    int b = g / PAD, i = g % PAD;
    float s = -2.0f;   // padding: strictly below the reference's -1 "invalid" score
    int idx = 0;
    if (i < NP) {
        const float* row = pred + ((size_t)b * NP + i) * NCOL;
        __builtin_prefetch(row + 64 * NCOL, 0, 1);   // global_prefetch_b8 stream-ahead
        float obj = row[4];
        float conf = obj * row[5];
        s = (obj > CONF_T && conf > CONF_T) ? conf : -1.0f;
        idx = i;
    }
    scores[g] = s;
    sidx[g] = idx;
}

__device__ __forceinline__ bool before(float sa, int ia, float sb, int ib) {
    // descending score, ascending index on ties (matches jax top_k stability)
    return (sa > sb) || (sa == sb && ia < ib);
}

// Global bitonic compare-exchange pass (only used for j >= 2048)
__global__ void bitonic_pass(float* __restrict__ scores, int* __restrict__ sidx,
                             int k, int j) {
    int g = blockIdx.x * blockDim.x + threadIdx.x;
    if (g >= NB * PAD) return;
    int b = g / PAD, i = g % PAD;
    int p = i ^ j;
    if (p <= i) return;
    size_t base = (size_t)b * PAD;
    float sa = scores[base + i], sb = scores[base + p];
    int   ia = sidx[base + i],   ib = sidx[base + p];
    bool wantDesc = ((i & k) == 0);
    bool aFirst = before(sa, ia, sb, ib);
    if (wantDesc ? !aFirst : aFirst) {
        scores[base + i] = sb; scores[base + p] = sa;
        sidx[base + i] = ib;   sidx[base + p] = ia;
    }
}

// Fused in-LDS bitonic stages: each block owns 2048 contiguous elements.
__global__ __launch_bounds__(1024) void bitonic_local(float* __restrict__ scores,
                                                      int* __restrict__ sidx,
                                                      int kstart, int kend) {
    __shared__ float ss[2048];
    __shared__ int   si[2048];
    int t = threadIdx.x;
    size_t base = (size_t)blockIdx.x * 2048;   // PAD % 2048 == 0: never crosses batches
    int gbase = (int)(base % PAD);
    ss[t] = scores[base + t];           si[t] = sidx[base + t];
    ss[t + 1024] = scores[base + t + 1024]; si[t + 1024] = sidx[base + t + 1024];
    __syncthreads();
    for (int k = kstart; k <= kend; k <<= 1) {
        int jstart = (k >> 1) < 1024 ? (k >> 1) : 1024;
        for (int j = jstart; j >= 1; j >>= 1) {
            int i = ((t & ~(j - 1)) << 1) | (t & (j - 1));
            int p = i + j;
            float sa = ss[i], sb = ss[p];
            int   ia = si[i], ib = si[p];
            bool wantDesc = (((gbase + i) & k) == 0);
            bool aFirst = before(sa, ia, sb, ib);
            if (wantDesc ? !aFirst : aFirst) {
                ss[i] = sb; ss[p] = sa; si[i] = ib; si[p] = ia;
            }
            __syncthreads();
        }
    }
    scores[base + t] = ss[t];           sidx[base + t] = si[t];
    scores[base + t + 1024] = ss[t + 1024]; sidx[base + t + 1024] = si[t + 1024];
}

// Phase 3: xywh -> xyxy for the sorted top-1024, init keep flags.
__global__ void prep_kernel(const float* __restrict__ pred,
                            const float* __restrict__ scores, const int* __restrict__ sidx,
                            float* __restrict__ boxesk, int* __restrict__ keep) {
    int g = blockIdx.x * blockDim.x + threadIdx.x;
    if (g >= NB * TOPK) return;
    int b = g / TOPK, t = g % TOPK;
    float s = scores[(size_t)b * PAD + t];
    int idx = sidx[(size_t)b * PAD + t];
    if (idx >= NP) idx = 0;
    const float* row = pred + ((size_t)b * NP + idx) * NCOL;
    float cx = row[0], cy = row[1], w = row[2], h = row[3];
    float* o = boxesk + (size_t)g * 4;
    o[0] = cx - w * 0.5f; o[1] = cy - h * 0.5f;
    o[2] = cx + w * 0.5f; o[3] = cy + h * 0.5f;
    keep[g] = (s > 0.0f) ? 1 : 0;
}

// Phase 4: greedy suppression. One block per batch. Boxes staged into LDS via the
// Tensor Data Mover (async DMA), completion via TENSORcnt.
__global__ __launch_bounds__(1024) void nms_kernel(const float* __restrict__ boxesk,
                                                   int* __restrict__ keep) {
    __shared__ float4 sbox[TOPK];   // 16 KB tile
    __shared__ int skeep[TOPK];
    int b = blockIdx.x, t = threadIdx.x;
    skeep[t] = keep[b * TOPK + t];

    if (t == 0) {
        uint64_t ga = (uint64_t)(uintptr_t)(boxesk + (size_t)b * TOPK * 4);
        uint32_t lds = (uint32_t)(uintptr_t)(&sbox[0]);   // LDS aperture: low 32 bits = offset
        v4u g0;                    // D# group 0: count=1, lds_addr, global_addr, type=2
        g0.x = 1u;
        g0.y = lds;
        g0.z = (uint32_t)ga;
        g0.w = ((uint32_t)(ga >> 32) & 0x01FFFFFFu) | (2u << 30);
        v8i g1;                    // data_size=4B, dim0=tile0=4096 dwords, 1 row
        g1[0] = 0x00020000;
        g1[1] = 0x10000000;
        g1[2] = 0x00010000;
        g1[3] = 0x10000000;
        g1[4] = 0x00000001;
        g1[5] = 4096;
        g1[6] = 0;
        g1[7] = 0;
        v4i g2 = {0, 0, 0, 0}, g3 = {0, 0, 0, 0};
#if __clang_major__ >= 23
        v8i g4 = {0, 0, 0, 0, 0, 0, 0, 0};
        __builtin_amdgcn_tensor_load_to_lds(g0, g1, g2, g3, g4, 0);
#else
        __builtin_amdgcn_tensor_load_to_lds(g0, g1, g2, g3, 0);
#endif
    }
    __builtin_amdgcn_s_wait_tensorcnt(0);   // waves with TENSORcnt==0 fall through
    __syncthreads();                        // publish LDS tile to all 32 waves

    float4 bj = sbox[t];
    float areaj = (bj.z - bj.x) * (bj.w - bj.y);
    for (int i = 0; i < TOPK - 1; ++i) {
        bool ki = (skeep[i] != 0);          // stable: iteration i only writes t > i
        if (ki && t > i && skeep[t]) {
            float4 bi = sbox[i];
            float ax = fmaxf(bi.x, bj.x), ay = fmaxf(bi.y, bj.y);
            float bx = fminf(bi.z, bj.z), by = fminf(bi.w, bj.w);
            float iw = fmaxf(bx - ax, 0.0f), ih = fmaxf(by - ay, 0.0f);
            float inter = iw * ih;
            float areai = (bi.z - bi.x) * (bi.w - bi.y);
            float iou = inter / (areai + areaj - inter + 1e-9f);
            if (iou > IOU_T) skeep[t] = 0;
        }
        __syncthreads();
    }
    keep[b * TOPK + t] = skeep[t];
}

// Phase 5: compact survivors. rank[t] = #kept before t, computed as a
// matmul-scan on the matrix core: flags as 32x32 f16 matrix F, within-row
// exclusive prefix P = F x U (U strictly-upper ones) via 4x
// V_WMMA_F32_16X16X32_F16 (exact: 0/1 values, sums <= 32 in f32).
__global__ __launch_bounds__(1024) void finalize_kernel(const float* __restrict__ pred,
                                                        const float* __restrict__ scores,
                                                        const int* __restrict__ sidx,
                                                        const float* __restrict__ boxesk,
                                                        const int* __restrict__ keep,
                                                        float* __restrict__ out) {
    __shared__ int skeep[TOPK];
    __shared__ float srank[TOPK];   // P[r][c], flag index t = 32*r + c
    __shared__ float rowoff[32];
    int b = blockIdx.x, t = threadIdx.x;
    skeep[t] = keep[b * TOPK + t];
    __syncthreads();

    if (t < 32) {                    // wave 0, EXEC all-ones: runs the WMMAs
        int L = t;
        int m = L & 15, hi = L >> 4;
        for (int h = 0; h < 2; ++h) {             // row halves: rows 16h..16h+15
            v16h a;                               // 16-bit A 16x32 lane layout
            #pragma unroll
            for (int e = 0; e < 16; ++e) {
                int K = (e < 8) ? (8 * hi + e) : (16 + 8 * hi + (e - 8));
                a[e] = (_Float16)(float)skeep[(16 * h + m) * 32 + K];
            }
            for (int nb = 0; nb < 2; ++nb) {      // column blocks of 16
                v16h bb;                          // 16-bit B 32x16 lane layout
                int c = 16 * nb + m;              // this lane's output column
                #pragma unroll
                for (int e = 0; e < 16; ++e) {
                    int K = (e < 8) ? (8 * hi + e) : (16 + 8 * hi + (e - 8));
                    bb[e] = (_Float16)((K < c) ? 1.0f : 0.0f);   // strictly-upper U
                }
                v8f cacc = {};
                cacc = __builtin_amdgcn_wmma_f32_16x16x32_f16(
                    /*neg_a=*/false, a, /*neg_b=*/false, bb,
                    /*c_mod=*/(short)0, cacc, /*reuse_a=*/false, /*reuse_b=*/false);
                #pragma unroll
                for (int v = 0; v < 8; ++v) {     // f32 C/D layout: M=v+8*hi, N=m
                    int M = v + 8 * hi;
                    srank[(16 * h + M) * 32 + c] = cacc[v];
                }
            }
        }
    }
    __syncthreads();
    if (t == 0) {                    // exclusive scan of 32 row totals
        float acc = 0.0f;
        for (int r = 0; r < 32; ++r) {
            rowoff[r] = acc;
            acc += srank[r * 32 + 31] + (float)skeep[r * 32 + 31];
        }
    }
    __syncthreads();

    if (!skeep[t]) return;
    int rank = (int)(rowoff[t >> 5] + srank[t]);
    if (rank >= MAXDET) return;
    float s = scores[(size_t)b * PAD + t];
    int idx = sidx[(size_t)b * PAD + t];
    const float* bx = boxesk + ((size_t)b * TOPK + t) * 4;
    float* row = out + ((size_t)b * MAXDET + rank) * NCOL;
    row[0] = bx[0]; row[1] = bx[1]; row[2] = bx[2]; row[3] = bx[3];
    row[4] = s;
    row[5] = 0.0f;   // NC==1 -> class always 0
    const float* pr = pred + ((size_t)b * NP + idx) * NCOL;
    #pragma unroll 3
    for (int q = 0; q < 51; ++q) row[6 + q] = pr[6 + q];
    out[(size_t)NB * MAXDET * NCOL + (size_t)b * MAXDET + rank] = 1.0f;
}

extern "C" void kernel_launch(void* const* d_in, const int* in_sizes, int n_in,
                              void* d_out, int out_size, void* d_ws, size_t ws_size,
                              hipStream_t stream) {
    (void)in_sizes; (void)n_in; (void)ws_size;
    const float* pred = (const float*)d_in[0];
    float* out = (float*)d_out;

    char* ws = (char*)d_ws;
    float* scores = (float*)(ws);                                  // NB*PAD f32  (2 MB)
    int*   sidx   = (int*)(ws + (size_t)NB * PAD * 4);             // NB*PAD i32  (2 MB)
    float* boxesk = (float*)(ws + (size_t)NB * PAD * 8);           // NB*TOPK*4   (256 KB)
    int*   keep   = (int*)(ws + (size_t)NB * PAD * 8 + (size_t)NB * TOPK * 16);

    zero_out<<<(out_size + 255) / 256, 256, 0, stream>>>(out, out_size);

    int total = NB * PAD;
    score_kernel<<<total / 256, 256, 0, stream>>>(pred, scores, sidx);

    int chunks = total / 2048;
    bitonic_local<<<chunks, 1024, 0, stream>>>(scores, sidx, 2, 2048);
    for (int k = 4096; k <= PAD; k <<= 1) {
        for (int j = k >> 1; j >= 2048; j >>= 1)
            bitonic_pass<<<total / 256, 256, 0, stream>>>(scores, sidx, k, j);
        bitonic_local<<<chunks, 1024, 0, stream>>>(scores, sidx, k, k);
    }

    prep_kernel<<<(NB * TOPK) / 256, 256, 0, stream>>>(pred, scores, sidx, boxesk, keep);
    nms_kernel<<<NB, 1024, 0, stream>>>(boxesk, keep);
    finalize_kernel<<<NB, 1024, 0, stream>>>(pred, scores, sidx, boxesk, keep, out);
}
